// Colla_AutoCorrelationLayer_19198503813440
// MI455X (gfx1250) — matrix operand order repaired and checked by reading, not compiled, not run on hardware
//
#include <hip/hip_runtime.h>
#include <cstdint>
#include <cstddef>

typedef __attribute__((ext_vector_type(16))) _Float16 v16h;
typedef __attribute__((ext_vector_type(8)))  float    v8f;

#define BB 2
#define LL 2048
#define SS 2048
#define DMODEL 512
#define HH 8
#define DH 64

struct H16u {
  union { v16h v; _Float16 h[16]; uint4 q[2]; };
  __device__ H16u() {}
};
struct H8u {
  union { _Float16 h[8]; uint4 q; };
  __device__ H8u() {}
};

__device__ __forceinline__ v8f wmma16x16x32(v16h a, v16h b, v8f c) {
  // D = A(16x32 f16) x B(32x16 f16) + C(16x16 f32)
  return __builtin_amdgcn_wmma_f32_16x16x32_f16(false, a, false, b, (short)0, c, false, false);
}

// 16 contiguous halves (B fragment: half i -> K = i (+16 for hi lanes), N = lane&15)
__device__ __forceinline__ v16h load16h_contig(const _Float16* p) {
  H16u t;
  t.q[0] = *(const uint4*)(p);
  t.q[1] = *(const uint4*)(p + 8);
  return t.v;
}

// A fragment from f16: halves 0-15 at p[0..15]
__device__ __forceinline__ v16h load16h_Asplit(const _Float16* p) {
  H16u t;
  t.q[0] = *(const uint4*)(p);
  t.q[1] = *(const uint4*)(p + 8);
  return t.v;
}

// A fragment from f32 source (convert to f16): 16 contiguous floats at p
__device__ __forceinline__ v16h cvt16h_Asplit_f32(const float* p) {
  float4 f0 = *(const float4*)(p);
  float4 f1 = *(const float4*)(p + 4);
  float4 f2 = *(const float4*)(p + 8);
  float4 f3 = *(const float4*)(p + 12);
  H16u t;
  t.h[0] = (_Float16)f0.x; t.h[1] = (_Float16)f0.y; t.h[2] = (_Float16)f0.z; t.h[3] = (_Float16)f0.w;
  t.h[4] = (_Float16)f1.x; t.h[5] = (_Float16)f1.y; t.h[6] = (_Float16)f1.z; t.h[7] = (_Float16)f1.w;
  t.h[8] = (_Float16)f2.x; t.h[9] = (_Float16)f2.y; t.h[10] = (_Float16)f2.z; t.h[11] = (_Float16)f2.w;
  t.h[12] = (_Float16)f3.x; t.h[13] = (_Float16)f3.y; t.h[14] = (_Float16)f3.z; t.h[15] = (_Float16)f3.w;
  return t.v;
}

// ---------------------------------------------------------------------------
// One-shot f32 -> f16 conversion, 8 elements/thread, b128 in / b128 out.
// ---------------------------------------------------------------------------
__global__ __launch_bounds__(256)
void cvt_f16(const float* __restrict__ src, _Float16* __restrict__ dst, int n) {
  int i = (blockIdx.x * 256 + threadIdx.x) * 8;
  if (i >= n) return;
  float4 a = *(const float4*)(src + i);
  float4 b = *(const float4*)(src + i + 4);
  H8u t;
  t.h[0] = (_Float16)a.x; t.h[1] = (_Float16)a.y; t.h[2] = (_Float16)a.z; t.h[3] = (_Float16)a.w;
  t.h[4] = (_Float16)b.x; t.h[5] = (_Float16)b.y; t.h[6] = (_Float16)b.z; t.h[7] = (_Float16)b.w;
  *(uint4*)(dst + i) = t.q;
}

// ---------------------------------------------------------------------------
// Y[M,N] (f16) = X[M,K] (f16) @ W[N,K]^T (f16) + bias(f32).  4 waves/block,
// each wave does a 16x64 strip (4 accumulators), A fragment reused 4x.
// Software-pipelined: next k-chunk's fragments are loaded before the WMMAs
// consuming the current chunk, so loads fly under the matrix pipe.
// grid = (M/16, N/256), block = 128.
// ---------------------------------------------------------------------------
__global__ __launch_bounds__(128)
void gemm_h16(const _Float16* __restrict__ X, const _Float16* __restrict__ W,
              const float* __restrict__ bias, _Float16* __restrict__ Y,
              int M, int N, int K) {
  const int wave = threadIdx.x >> 5;
  const int lane = threadIdx.x & 31;
  const int m0   = blockIdx.x << 4;
  const int nb   = (blockIdx.y << 8) + (wave << 6);
  const int mrow = lane & 15;
  const int hi   = lane >> 4;

  const v8f vzero = {0.f, 0.f, 0.f, 0.f, 0.f, 0.f, 0.f, 0.f};
  v8f acc0 = vzero, acc1 = vzero, acc2 = vzero, acc3 = vzero;

  const _Float16* Xrow = X + (size_t)(m0 + mrow) * K + hi * 16;
  const _Float16* Wr0  = W + (size_t)(nb + mrow) * K + hi * 16;
  const _Float16* Wr1  = Wr0 + (size_t)16 * K;
  const _Float16* Wr2  = Wr0 + (size_t)32 * K;
  const _Float16* Wr3  = Wr0 + (size_t)48 * K;

  v16h a  = load16h_Asplit(Xrow);
  v16h b0 = load16h_contig(Wr0);
  v16h b1 = load16h_contig(Wr1);
  v16h b2 = load16h_contig(Wr2);
  v16h b3 = load16h_contig(Wr3);
  for (int kc = 32; kc < K; kc += 32) {
    v16h an  = load16h_Asplit(Xrow + kc);
    v16h bn0 = load16h_contig(Wr0 + kc);
    v16h bn1 = load16h_contig(Wr1 + kc);
    v16h bn2 = load16h_contig(Wr2 + kc);
    v16h bn3 = load16h_contig(Wr3 + kc);
    acc0 = wmma16x16x32(a, b0, acc0);
    acc1 = wmma16x16x32(a, b1, acc1);
    acc2 = wmma16x16x32(a, b2, acc2);
    acc3 = wmma16x16x32(a, b3, acc3);
    a = an; b0 = bn0; b1 = bn1; b2 = bn2; b3 = bn3;
  }
  acc0 = wmma16x16x32(a, b0, acc0);
  acc1 = wmma16x16x32(a, b1, acc1);
  acc2 = wmma16x16x32(a, b2, acc2);
  acc3 = wmma16x16x32(a, b3, acc3);

  v8f accs[4] = {acc0, acc1, acc2, acc3};
#pragma unroll
  for (int t = 0; t < 4; ++t) {
    int n = nb + (t << 4) + mrow;
    float bia = bias ? bias[n] : 0.0f;
#pragma unroll
    for (int r = 0; r < 8; ++r) {
      int mm = m0 + r + (hi << 3);
      Y[(size_t)mm * N + n] = (_Float16)(accs[t][r] + bia);
    }
  }
}

// Same, but f32 output (final projection writes directly into d_out).
__global__ __launch_bounds__(128)
void gemm_h32(const _Float16* __restrict__ X, const _Float16* __restrict__ W,
              const float* __restrict__ bias, float* __restrict__ Y,
              int M, int N, int K) {
  const int wave = threadIdx.x >> 5;
  const int lane = threadIdx.x & 31;
  const int m0   = blockIdx.x << 4;
  const int nb   = (blockIdx.y << 8) + (wave << 6);
  const int mrow = lane & 15;
  const int hi   = lane >> 4;

  const v8f vzero = {0.f, 0.f, 0.f, 0.f, 0.f, 0.f, 0.f, 0.f};
  v8f acc0 = vzero, acc1 = vzero, acc2 = vzero, acc3 = vzero;

  const _Float16* Xrow = X + (size_t)(m0 + mrow) * K + hi * 16;
  const _Float16* Wr0  = W + (size_t)(nb + mrow) * K + hi * 16;
  const _Float16* Wr1  = Wr0 + (size_t)16 * K;
  const _Float16* Wr2  = Wr0 + (size_t)32 * K;
  const _Float16* Wr3  = Wr0 + (size_t)48 * K;

  v16h a  = load16h_Asplit(Xrow);
  v16h b0 = load16h_contig(Wr0);
  v16h b1 = load16h_contig(Wr1);
  v16h b2 = load16h_contig(Wr2);
  v16h b3 = load16h_contig(Wr3);
  for (int kc = 32; kc < K; kc += 32) {
    v16h an  = load16h_Asplit(Xrow + kc);
    v16h bn0 = load16h_contig(Wr0 + kc);
    v16h bn1 = load16h_contig(Wr1 + kc);
    v16h bn2 = load16h_contig(Wr2 + kc);
    v16h bn3 = load16h_contig(Wr3 + kc);
    acc0 = wmma16x16x32(a, b0, acc0);
    acc1 = wmma16x16x32(a, b1, acc1);
    acc2 = wmma16x16x32(a, b2, acc2);
    acc3 = wmma16x16x32(a, b3, acc3);
    a = an; b0 = bn0; b1 = bn1; b2 = bn2; b3 = bn3;
  }
  acc0 = wmma16x16x32(a, b0, acc0);
  acc1 = wmma16x16x32(a, b1, acc1);
  acc2 = wmma16x16x32(a, b2, acc2);
  acc3 = wmma16x16x32(a, b3, acc3);

  v8f accs[4] = {acc0, acc1, acc2, acc3};
#pragma unroll
  for (int t = 0; t < 4; ++t) {
    int n = nb + (t << 4) + mrow;
    float bia = bias ? bias[n] : 0.0f;
#pragma unroll
    for (int r = 0; r < 8; ++r) {
      int mm = m0 + r + (hi << 3);
      Y[(size_t)mm * N + n] = accs[t][r] + bia;
    }
  }
}

// ---------------------------------------------------------------------------
// cb[b,h,s] = keys[b,s,:] . Wcb[h,:]  in f32 (tiny; keeps bias path precise)
// ---------------------------------------------------------------------------
__global__ __launch_bounds__(256)
void cb_kernel(const float* __restrict__ keys, const float* __restrict__ Wcb,
               float* __restrict__ cb) {
  int idx = blockIdx.x * 256 + threadIdx.x;      // ((b*H + h)*S + s)
  if (idx >= BB * HH * SS) return;
  int s  = idx & (SS - 1);
  int bh = idx >> 11;
  int h  = bh & (HH - 1);
  int b  = bh >> 3;
  const float* krow = keys + (size_t)(b * SS + s) * DMODEL;
  const float* wrow = Wcb + (size_t)h * DMODEL;
  float accv = 0.f;
  for (int i = 0; i < DMODEL; i += 4) {
    float4 kv = *(const float4*)(krow + i);
    float4 wv = *(const float4*)(wrow + i);
    accv += kv.x * wv.x + kv.y * wv.y + kv.z * wv.z + kv.w * wv.w;
  }
  cb[idx] = accv;
}

// ---------------------------------------------------------------------------
// vT[(b*H+h)*64 + dv][s] = v[b][s][h*64+dv]   (LDS-tiled 64x64 transpose,
// coalesced on both sides; makes ctx B-fragments contiguous b128 loads)
// grid = BB*HH*(SS/64), block = 256
// ---------------------------------------------------------------------------
__global__ __launch_bounds__(256)
void vtrans_kernel(const _Float16* __restrict__ vf, _Float16* __restrict__ vT) {
  __shared__ _Float16 tile[64 * 65];
  const int bid = blockIdx.x;          // ((b*H + h)*32 + st)
  const int st  = bid & 31;
  const int bh  = bid >> 5;
  const int h   = bh & (HH - 1);
  const int b   = bh >> 3;
  const int s0  = st << 6;
  const int tid = threadIdx.x;
  for (int i = tid; i < 64 * 64; i += 256) {
    int sl = i >> 6, dvl = i & 63;
    tile[sl * 65 + dvl] = vf[(size_t)(b * SS + s0 + sl) * DMODEL + (h << 6) + dvl];
  }
  __syncthreads();
  for (int i = tid; i < 64 * 64; i += 256) {
    int dvl = i >> 6, sl = i & 63;
    vT[(size_t)((bh << 6) + dvl) * SS + s0 + sl] = tile[sl * 65 + dvl];
  }
}

// ---------------------------------------------------------------------------
// Fused attention: one block = one (b, h, 16-row l-tile).
//   phase 0: mixed-q (q * mixing_h, f16) -> LDS (16 KB)
//   phase 1: scores[16 x 2048] via WMMA (2 s-tiles share each A frag,
//            double-buffered fragments) -> LDS f32
//   phase 2: softmax in LDS, normalized probs -> d_out (b128 stores)
//   phase 3: ctx[16 x 64] = probs @ vT via WMMA (double-buffered)
// 145 KB LDS/workgroup (CDNA5: 320 KB/WGP).
// ---------------------------------------------------------------------------
__global__ __launch_bounds__(256)
void attn_kernel(const _Float16* __restrict__ qf, const _Float16* __restrict__ kf,
                 const _Float16* __restrict__ vT, const float* __restrict__ cbp,
                 const float* __restrict__ mixing,
                 float* __restrict__ probs_out, _Float16* __restrict__ ctx_out) {
  __shared__ _Float16 mq[16 * DMODEL];     // 16 KB
  __shared__ float    sc[16 * SS];         // 128 KB
  __shared__ float    redbuf[256];
  __shared__ float    rowmax[16];
  __shared__ float    rowinv[16];

  const int bid  = blockIdx.x;             // ((b*H + h)*128 + lt)
  const int lt   = bid & 127;
  const int bh   = bid >> 7;
  const int h    = bh & (HH - 1);
  const int b    = bh >> 3;
  const int l0   = lt << 4;
  const int tid  = threadIdx.x;
  const int wave = tid >> 5;
  const int lane = tid & 31;
  const int mrow = lane & 15;
  const int hi   = lane >> 4;
  const v8f vzero = {0.f, 0.f, 0.f, 0.f, 0.f, 0.f, 0.f, 0.f};

  // ---- phase 0: stage mixed-q tile (f16) into LDS ----
  for (int i = tid; i < 16 * DMODEL; i += 256) {
    int r = i >> 9;
    int e = i & (DMODEL - 1);
    float qv = (float)qf[(size_t)(b * LL + l0 + r) * DMODEL + e];
    mq[i] = (_Float16)(qv * mixing[h * DMODEL + e]);
  }
  __syncthreads();

  // ---- phase 1: scores; wave handles pairs of s-tiles sharing the A frag ----
  const float* cbrow = cbp + (size_t)(b * HH + h) * SS;
  const _Float16* mqrow = &mq[mrow * DMODEL + hi * 16];
  for (int stp = wave; stp < SS / 32; stp += 8) {
    const int s0 = stp << 5;
    v8f acc0 = vzero, acc1 = vzero;
    const _Float16* krow0 = kf + (size_t)(b * SS + s0 + mrow) * DMODEL + hi * 16;
    const _Float16* krow1 = krow0 + (size_t)16 * DMODEL;

    v16h a  = load16h_Asplit(mqrow);
    v16h b0 = load16h_contig(krow0);
    v16h b1 = load16h_contig(krow1);
    for (int kc = 32; kc < DMODEL; kc += 32) {
      v16h an  = load16h_Asplit(mqrow + kc);
      v16h bn0 = load16h_contig(krow0 + kc);
      v16h bn1 = load16h_contig(krow1 + kc);
      acc0 = wmma16x16x32(a, b0, acc0);
      acc1 = wmma16x16x32(a, b1, acc1);
      a = an; b0 = bn0; b1 = bn1;
    }
    acc0 = wmma16x16x32(a, b0, acc0);
    acc1 = wmma16x16x32(a, b1, acc1);

    float cb0 = cbrow[s0 + mrow];
    float cb1 = cbrow[s0 + 16 + mrow];
#pragma unroll
    for (int r = 0; r < 8; ++r) {
      sc[(r + (hi << 3)) * SS + s0 + mrow]      = (acc0[r] + cb0) * 0.125f;
      sc[(r + (hi << 3)) * SS + s0 + 16 + mrow] = (acc1[r] + cb1) * 0.125f;
    }
  }
  __syncthreads();

  // ---- phase 2a: per-row max and exp-sum (16 threads per row) ----
  {
    const int r = tid >> 4;
    const int c = tid & 15;
    float mx = -1e30f;
    for (int s = c; s < SS; s += 16) mx = fmaxf(mx, sc[r * SS + s]);
    redbuf[tid] = mx;
    __syncthreads();
    if (c == 0) {
      float v = redbuf[r * 16];
      for (int j = 1; j < 16; ++j) v = fmaxf(v, redbuf[r * 16 + j]);
      rowmax[r] = v;
    }
    __syncthreads();
    const float rm = rowmax[r];
    float sm = 0.f;
    for (int s = c; s < SS; s += 16) {
      float e = __expf(sc[r * SS + s] - rm);
      sc[r * SS + s] = e;
      sm += e;
    }
    redbuf[tid] = sm;
    __syncthreads();
    if (c == 0) {
      float v = 0.f;
      for (int j = 0; j < 16; ++j) v += redbuf[r * 16 + j];
      rowinv[r] = 1.0f / v;
    }
    __syncthreads();
  }

  // ---- phase 2b: normalize in LDS + b128 probs store ----
  const size_t pbase = ((size_t)(b * HH + h) * LL + l0) * SS;
  for (int r = 0; r < 16; ++r) {
    const float inv = rowinv[r];
    for (int s4 = tid * 4; s4 < SS; s4 += 1024) {
      float4 pv = *(float4*)&sc[r * SS + s4];
      pv.x *= inv; pv.y *= inv; pv.z *= inv; pv.w *= inv;
      *(float4*)&sc[r * SS + s4] = pv;
      *(float4*)(probs_out + pbase + (size_t)r * SS + s4) = pv;
    }
  }
  __syncthreads();

  // ---- phase 3: ctx[16 x 64] = probs @ v_head (waves 0..3, one 16-col tile each) ----
  if (wave < 4) {
    v8f acc = vzero;
    const int dv = (wave << 4) + mrow;     // column within head (0..63)
    const _Float16* vrow = vT + (size_t)((bh << 6) + dv) * SS + hi * 16;
    const float* prow = &sc[mrow * SS + hi * 16];

    v16h a  = cvt16h_Asplit_f32(prow);
    v16h bf = load16h_contig(vrow);
    for (int kc = 32; kc < SS; kc += 32) {
      v16h an  = cvt16h_Asplit_f32(prow + kc);
      v16h bfn = load16h_contig(vrow + kc);
      acc = wmma16x16x32(a, bf, acc);
      a = an; bf = bfn;
    }
    acc = wmma16x16x32(a, bf, acc);
#pragma unroll
    for (int r = 0; r < 8; ++r) {
      int mm = r + (hi << 3);
      ctx_out[(size_t)(b * LL + l0 + mm) * DMODEL + (h << 6) + dv] = (_Float16)acc[r];
    }
  }
}

// ---------------------------------------------------------------------------
extern "C" void kernel_launch(void* const* d_in, const int* in_sizes, int n_in,
                              void* d_out, int out_size, void* d_ws, size_t ws_size,
                              hipStream_t stream) {
  (void)in_sizes; (void)n_in; (void)out_size; (void)ws_size;
  const float* queries = (const float*)d_in[0];
  const float* keys    = (const float*)d_in[1];
  const float* values  = (const float*)d_in[2];
  // d_in[3] = attn_mask (unused, == 0 in reference)
  const float* Wq  = (const float*)d_in[4];
  const float* Wk  = (const float*)d_in[5];
  const float* Wv  = (const float*)d_in[6];
  const float* bv  = (const float*)d_in[7];
  const float* Wcb = (const float*)d_in[8];
  const float* mix = (const float*)d_in[9];
  const float* Wd  = (const float*)d_in[10];
  const float* bd  = (const float*)d_in[11];

  float* out   = (float*)d_out;
  float* probs = out + (size_t)BB * LL * DMODEL;

  const size_t MB = 1024 * 1024;
  char* w = (char*)d_ws;
  _Float16* xq16  = (_Float16*)(w + 0 * MB);    // f16 copies of activations
  _Float16* xk16  = (_Float16*)(w + 4 * MB);
  _Float16* xv16  = (_Float16*)(w + 8 * MB);
  _Float16* qf16  = (_Float16*)(w + 12 * MB);   // projected q/k/v
  _Float16* kf16  = (_Float16*)(w + 16 * MB);
  _Float16* vf16  = (_Float16*)(w + 20 * MB);
  _Float16* vT16  = (_Float16*)(w + 24 * MB);   // v transposed per head
  _Float16* ctx16 = (_Float16*)(w + 28 * MB);
  _Float16* Wq16  = (_Float16*)(w + 32 * MB);   // f16 weights (512 KB each)
  _Float16* Wk16  = (_Float16*)(w + 32 * MB + 512 * 1024);
  _Float16* Wv16  = (_Float16*)(w + 33 * MB);
  _Float16* Wd16  = (_Float16*)(w + 33 * MB + 512 * 1024);
  float*    cbw   = (float*)   (w + 34 * MB);

  const int M     = BB * LL;                 // 4096
  const int NACT  = M * DMODEL;              // 2,097,152
  const int NWGT  = DMODEL * DMODEL;         // 262,144
  dim3 ggrid(M / 16, DMODEL / 256);

  // one-shot f32 -> f16 conversions (activations + weights)
  cvt_f16<<<NACT / (8 * 256), 256, 0, stream>>>(queries, xq16, NACT);
  cvt_f16<<<NACT / (8 * 256), 256, 0, stream>>>(keys,    xk16, NACT);
  cvt_f16<<<NACT / (8 * 256), 256, 0, stream>>>(values,  xv16, NACT);
  cvt_f16<<<NWGT / (8 * 256), 256, 0, stream>>>(Wq, Wq16, NWGT);
  cvt_f16<<<NWGT / (8 * 256), 256, 0, stream>>>(Wk, Wk16, NWGT);
  cvt_f16<<<NWGT / (8 * 256), 256, 0, stream>>>(Wv, Wv16, NWGT);
  cvt_f16<<<NWGT / (8 * 256), 256, 0, stream>>>(Wd, Wd16, NWGT);

  // projections (pure f16 WMMA GEMMs)
  gemm_h16<<<ggrid, 128, 0, stream>>>(xq16, Wq16, nullptr, qf16, M, DMODEL, DMODEL);
  gemm_h16<<<ggrid, 128, 0, stream>>>(xk16, Wk16, nullptr, kf16, M, DMODEL, DMODEL);
  gemm_h16<<<ggrid, 128, 0, stream>>>(xv16, Wv16, bv,      vf16, M, DMODEL, DMODEL);

  cb_kernel<<<(BB * HH * SS) / 256, 256, 0, stream>>>(keys, Wcb, cbw);
  vtrans_kernel<<<BB * HH * (SS / 64), 256, 0, stream>>>(vf16, vT16);

  attn_kernel<<<BB * HH * (LL / 16), 256, 0, stream>>>(qf16, kf16, vT16, cbw, mix, probs, ctx16);

  gemm_h32<<<ggrid, 128, 0, stream>>>(ctx16, Wd16, bd, out, M, DMODEL, DMODEL);
}
